// NeuralMemoryBank_62680752718465
// MI455X (gfx1250) — matrix operand order repaired and checked by reading, not compiled, not run on hardware
//
#include <hip/hip_runtime.h>

// ---- CDNA5 / gfx1250 vector types ----
typedef __attribute__((ext_vector_type(16))) __bf16 v16bf;
typedef __attribute__((ext_vector_type(8)))  __bf16 v8bf;
typedef __attribute__((ext_vector_type(4)))  __bf16 v4bf;
typedef __attribute__((ext_vector_type(8)))  float  v8f;

#define NB 8
#define NS 4096
#define NH 256
#define NM 4096
#define NR (NB * NS)        // 32768 query rows
#define NWAVES 4            // waves per workgroup
#define WGROWS 64           // query rows per workgroup (16 per wave)
#define MC 32               // memory rows per chunk (matches bf16 WMMA K=32)

// Convert 8 contiguous f32 (global) to bf16 half-fragment.
__device__ __forceinline__ v8bf cvt8g(const float* __restrict__ p) {
  v8bf r;
#pragma unroll
  for (int i = 0; i < 8; ++i) r[i] = (__bf16)p[i];
  return r;
}

// Assemble a v16bf WMMA fragment from two 16-byte LDS chunks.
__device__ __forceinline__ v16bf ld_frag(const __bf16* p0, const __bf16* p1) {
  union { v16bf v; v8bf h[2]; } u;
  u.h[0] = *(const v8bf*)p0;
  u.h[1] = *(const v8bf*)p1;
  return u.v;
}

// =====================================================================
// Fused flash kernel: scores = x·memT, online softmax, read = P·mem,
// plus gate (sigmoid-sum over rows) and read-column-sum side reductions.
// =====================================================================
__global__ __launch_bounds__(128) void nmb_flash_kernel(
    const float* __restrict__ x, const float* __restrict__ mem,
    float* __restrict__ outRead, float* __restrict__ gGate,
    float* __restrict__ gRead)
{
  __shared__ __align__(16) __bf16 sMem[MC * NH];        // 32x256 row-major   (scores B)
  __shared__ __align__(16) __bf16 sMemT[NH * MC];       // 256x32 transposed  (read   B)
  __shared__ __align__(16) __bf16 sP[NWAVES][16 * MC];  // per-wave P scratch (C->A relayout)
  __shared__ float sGate[NM];
  __shared__ float sRead[NH];

  const int tid = threadIdx.x;
  const int w   = tid >> 5;          // wave id in WG
  const int l   = tid & 31;          // lane
  const int q0  = blockIdx.x * WGROWS + w * 16;   // this wave's first query row

  for (int i = tid; i < NM; i += 128) sGate[i] = 0.0f;
  for (int i = tid; i < NH; i += 128) sRead[i] = 0.0f;

  const int rowA   = l & 15;               // A row / B column / C column index
  const int hiHalf = (l >> 4) & 1;         // lane half
  const int koffA  = hiHalf ? 8 : 0;       // 16-bit A layout: K base per lane half
  const int koffB  = hiHalf ? 16 : 0;      // unused; B loads use koffA (same K order as A)

  // Persistent x tile: 16 rows x 256, as 8 bf16 A-fragments (K=32 each).
  // A layout (16-bit, 16x32): lanes<16 hold K={kb..kb+7, kb+16..kb+23},
  // lanes>=16 hold K={kb+8..kb+15, kb+24..kb+31}.
  v16bf Ax[8];
#pragma unroll
  for (int f = 0; f < 8; ++f) {
    const float* bp = x + (q0 + rowA) * NH + f * 32 + koffA;
    union { v16bf v; v8bf h[2]; } u;
    u.h[0] = cvt8g(bp);
    u.h[1] = cvt8g(bp + 16);
    Ax[f] = u.v;
  }

  // O accumulator: 16 C-tiles covering h = 0..255 for this wave's 16 rows.
  v8f acc[16];
#pragma unroll
  for (int t = 0; t < 16; ++t)
#pragma unroll
    for (int i = 0; i < 8; ++i) acc[t][i] = 0.0f;

  float rm[8], ls[8];   // running row max / row sum (row = v + 8*hiHalf)
#pragma unroll
  for (int v = 0; v < 8; ++v) { rm[v] = -3.0e38f; ls[v] = 0.0f; }

  for (int mb = 0; mb < NM; mb += MC) {
    __syncthreads();
    // ---- cooperative stage: 32x256 f32 memory tile -> bf16 (row-major + transposed)
#pragma unroll
    for (int it = 0; it < 16; ++it) {
      const int e = tid * 4 + it * 512;
      const int k = e >> 8;     // m-local row 0..31
      const int h = e & 255;    // h column (multiple of 4)
      float4 d = *(const float4*)(mem + (mb + k) * NH + h);
      v4bf pk;
      pk[0] = (__bf16)d.x; pk[1] = (__bf16)d.y;
      pk[2] = (__bf16)d.z; pk[3] = (__bf16)d.w;
      *(v4bf*)(&sMem[k * NH + h]) = pk;
      sMemT[(h + 0) * MC + k] = pk[0];
      sMemT[(h + 1) * MC + k] = pk[1];
      sMemT[(h + 2) * MC + k] = pk[2];
      sMemT[(h + 3) * MC + k] = pk[3];
    }
    if (mb + MC < NM)  // global_prefetch_b8 of next chunk
      __builtin_prefetch(mem + (size_t)(mb + MC) * NH + tid * 4, 0, 3);
    __syncthreads();

    // ---- scores: two 16x16 tiles (m-cols mb..mb+31), K-loop over H=256
    v8f c0, c1;
#pragma unroll
    for (int i = 0; i < 8; ++i) { c0[i] = 0.0f; c1[i] = 0.0f; }
#pragma unroll
    for (int f = 0; f < 8; ++f) {
      v16bf a = Ax[f];
      const __bf16* b0p = &sMem[rowA * NH + f * 32 + koffA];
      v16bf b0 = ld_frag(b0p, b0p + 16);
      c0 = __builtin_amdgcn_wmma_f32_16x16x32_bf16(false, a, false, b0,
                                                   (short)0, c0, false, false);
      const __bf16* b1p = &sMem[(16 + rowA) * NH + f * 32 + koffA];
      v16bf b1 = ld_frag(b1p, b1p + 16);
      c1 = __builtin_amdgcn_wmma_f32_16x16x32_bf16(false, a, false, b1,
                                                   (short)0, c1, false, false);
    }

    // ---- online softmax update + gate accumulation + P store (bf16)
    float g0 = 0.0f, g1 = 0.0f;
#pragma unroll
    for (int v = 0; v < 8; ++v) {
      const float s0 = c0[v];
      const float s1 = c1[v];
      g0 += 1.0f / (1.0f + __expf(-s0));
      g1 += 1.0f / (1.0f + __expf(-s1));
      float mx = fmaxf(s0, s1);                    // row max across 16 lanes
      mx = fmaxf(mx, __shfl_xor(mx, 1));
      mx = fmaxf(mx, __shfl_xor(mx, 2));
      mx = fmaxf(mx, __shfl_xor(mx, 4));
      mx = fmaxf(mx, __shfl_xor(mx, 8));
      const float nmax = fmaxf(rm[v], mx);
      const float sc = __expf(rm[v] - nmax);
      rm[v] = nmax;
      const float p0 = __expf(s0 - nmax);
      const float p1 = __expf(s1 - nmax);
      float rs = p0 + p1;
      rs += __shfl_xor(rs, 1);
      rs += __shfl_xor(rs, 2);
      rs += __shfl_xor(rs, 4);
      rs += __shfl_xor(rs, 8);
      ls[v] = ls[v] * sc + rs;
#pragma unroll
      for (int t = 0; t < 16; ++t) acc[t][v] *= sc; // rescale O
      const int prow = v + (hiHalf ? 8 : 0);
      sP[w][prow * MC + rowA]      = (__bf16)p0;    // row-major 16x32 P
      sP[w][prow * MC + 16 + rowA] = (__bf16)p1;
    }
    // gate: finish column sums over all 16 rows, LDS-atomic accumulate
    g0 += __shfl_xor(g0, 16);
    g1 += __shfl_xor(g1, 16);
    if (l < 16) {
      atomicAdd(&sGate[mb + l],      g0);
      atomicAdd(&sGate[mb + 16 + l], g1);
    }
    asm volatile("s_wait_dscnt 0" ::: "memory");    // P stores -> P loads (same wave)

    // ---- O += P(16x32) x memChunk(32x256): 16 independent WMMA chains
    const __bf16* pp = &sP[w][rowA * MC + koffA];
    v16bf pa = ld_frag(pp, pp + 16);
#pragma unroll
    for (int t = 0; t < 16; ++t) {
      const __bf16* bp = &sMemT[(t * 16 + rowA) * MC + koffA];
      v16bf b = ld_frag(bp, bp + 16);
      acc[t] = __builtin_amdgcn_wmma_f32_16x16x32_bf16(false, pa, false, b,
                                                       (short)0, acc[t], false, false);
    }
  }

  // ---- finalize: read = O / l ; store + column sums for read_avg
  float rinv[8];
#pragma unroll
  for (int v = 0; v < 8; ++v) rinv[v] = 1.0f / ls[v];

#pragma unroll
  for (int t = 0; t < 16; ++t) {
    float colsum = 0.0f;
#pragma unroll
    for (int v = 0; v < 8; ++v) {
      const float val = acc[t][v] * rinv[v];
      const int row = v + (hiHalf ? 8 : 0);
      outRead[(q0 + row) * NH + t * 16 + rowA] = val;
      colsum += val;
    }
    colsum += __shfl_xor(colsum, 16);
    if (l < 16) atomicAdd(&sRead[t * 16 + l], colsum);
  }

  __syncthreads();
  for (int i = tid; i < NM; i += 128) atomicAdd(&gGate[i], sGate[i]);
  for (int i = tid; i < NH; i += 128) atomicAdd(&gRead[i], sRead[i]);
}

// =====================================================================
// Epilogue: gated memory update + layernorm over H=256 per memory row.
// =====================================================================
__global__ __launch_bounds__(256) void nmb_epilogue_kernel(
    const float* __restrict__ mem, const float* __restrict__ gamma,
    const float* __restrict__ beta, const float* __restrict__ gGate,
    const float* __restrict__ gRead, float* __restrict__ outMem)
{
  __shared__ float red[256];
  const int m = blockIdx.x;
  const int h = threadIdx.x;
  const float inv_r = 1.0f / (float)NR;
  const float g = gGate[m] * inv_r;                 // mean sigmoid(scores) for row m
  const float ravg = gRead[h] * inv_r;              // mean read for col h
  const float val = mem[m * NH + h] * (1.0f - g) + ravg * g;

  red[h] = val;
  __syncthreads();
  for (int off = 128; off > 0; off >>= 1) {
    if (h < off) red[h] += red[h + off];
    __syncthreads();
  }
  const float mu = red[0] * (1.0f / (float)NH);
  __syncthreads();
  const float d = val - mu;
  red[h] = d * d;
  __syncthreads();
  for (int off = 128; off > 0; off >>= 1) {
    if (h < off) red[h] += red[h + off];
    __syncthreads();
  }
  const float var = red[0] * (1.0f / (float)NH);
  outMem[m * NH + h] = d * rsqrtf(var + 1e-5f) * gamma[h] + beta[h];
}

__global__ void nmb_zero_kernel(float* __restrict__ ws, int n) {
  const int i = blockIdx.x * blockDim.x + threadIdx.x;
  if (i < n) ws[i] = 0.0f;
}

extern "C" void kernel_launch(void* const* d_in, const int* in_sizes, int n_in,
                              void* d_out, int out_size, void* d_ws, size_t ws_size,
                              hipStream_t stream) {
  (void)in_sizes; (void)n_in; (void)out_size; (void)ws_size;
  const float* x     = (const float*)d_in[0];   // (8,4096,256)
  const float* mem   = (const float*)d_in[1];   // (4096,256)
  const float* gamma = (const float*)d_in[2];   // (256,)
  const float* beta  = (const float*)d_in[3];   // (256,)

  float* outRead = (float*)d_out;                       // (8,4096,256) flat
  float* outMem  = (float*)d_out + (size_t)NR * NH;     // (4096,256)   flat
  float* gGate   = (float*)d_ws;                        // NM f32 sigmoid-sum
  float* gRead   = gGate + NM;                          // NH f32 read-sum

  nmb_zero_kernel<<<(NM + NH + 255) / 256, 256, 0, stream>>>(gGate, NM + NH);
  nmb_flash_kernel<<<NR / WGROWS, 128, 0, stream>>>(x, mem, outRead, gGate, gRead);
  nmb_epilogue_kernel<<<NM, 256, 0, stream>>>(mem, gamma, beta, gGate, gRead, outMem);
}
